// MoE_HDM_46205258171030
// MI455X (gfx1250) — matrix operand order repaired and checked by reading, not compiled, not run on hardware
//
#include <hip/hip_runtime.h>
#include <math.h>

// MoE forward for MI455X (gfx1250, wave32).
// Fast path: token-sorted sparse expert GEMM via split-precision f16 WMMA
//   out = x_hi*w16 + x_lo*w16   (x split compensates A rounding; W f16)
// using V_WMMA_F32_16X16X32_F16. Each wave computes two 16x16 output tiles,
// reusing the A operand -> 1 ds_load_b128 + 1 global_load_b128 per WMMA.
// w_exp is pre-packed to 64MB f16 (L2-resident; 192MB L2) in a B-operand
// lane-major tiling (contiguous k per lane half) so each wave's B fetch is contiguous.
// Fallback path (if d_ws can't hold the packed W): exact-f32 WMMA x4 kernel.

#define BB 4096
#define DD 1024
#define EE 8
#define CC 4096
#define KTOP 4
#define TNF 256        // f16 kernel: columns per block = 8 waves * 2 tiles * 16
#define TN 128         // f32 fallback: columns per block = 8 waves * 16
#define KC 256         // K-chunk staged in LDS (f32 fallback kernel)
#define XSTRIDE 1032   // f16 LDS row stride (1024 + 8 pad)
#define EPS_COMBINE 2.220446049250313e-16f

typedef float    v2f  __attribute__((ext_vector_type(2)));
typedef float    v8f  __attribute__((ext_vector_type(8)));
typedef _Float16 v4h  __attribute__((ext_vector_type(4)));
typedef _Float16 v8h  __attribute__((ext_vector_type(8)));
typedef _Float16 v16h __attribute__((ext_vector_type(16)));

// ---------------- Kernel 1: gating (logits, top-4, softmax, mask) ----------
__global__ __launch_bounds__(128) void gating_kernel(
    const float* __restrict__ x, const float* __restrict__ wg,
    float* __restrict__ gate4, int* __restrict__ idx4, int* __restrict__ maskb) {
  int b = blockIdx.x * blockDim.x + threadIdx.x;
  if (b >= BB) return;
  const float* xr = x + (size_t)b * DD;
  float acc[EE];
#pragma unroll
  for (int e = 0; e < EE; ++e) acc[e] = 0.f;
  for (int d = 0; d < DD; d += 4) {
    float4 xv = *(const float4*)(xr + d);
    float xq[4] = {xv.x, xv.y, xv.z, xv.w};
#pragma unroll
    for (int q = 0; q < 4; ++q) {
      const float4* wr = (const float4*)(wg + (size_t)(d + q) * EE);
      float4 wa = wr[0], wb = wr[1];
      acc[0] += xq[q] * wa.x; acc[1] += xq[q] * wa.y;
      acc[2] += xq[q] * wa.z; acc[3] += xq[q] * wa.w;
      acc[4] += xq[q] * wb.x; acc[5] += xq[q] * wb.y;
      acc[6] += xq[q] * wb.z; acc[7] += xq[q] * wb.w;
    }
  }
  float tmp[EE];
#pragma unroll
  for (int e = 0; e < EE; ++e) tmp[e] = acc[e];
  int pick[KTOP]; float val[KTOP];
#pragma unroll
  for (int s = 0; s < KTOP; ++s) {
    int bi = 0; float bv = tmp[0];
#pragma unroll
    for (int j = 1; j < EE; ++j) { if (tmp[j] > bv) { bv = tmp[j]; bi = j; } }
    pick[s] = bi; val[s] = bv; tmp[bi] = -INFINITY;
  }
  float m = val[0], ssum = 0.f, ex[KTOP];
#pragma unroll
  for (int s = 0; s < KTOP; ++s) { ex[s] = __expf(val[s] - m); ssum += ex[s]; }
  int msk = 0;
#pragma unroll
  for (int s = 0; s < KTOP; ++s) {
    gate4[b * KTOP + s] = ex[s] / ssum;
    idx4[b * KTOP + s] = pick[s];
    msk |= 1 << pick[s];
  }
  maskb[b] = msk;
}

// ---------------- Kernel 2: aux loss (deterministic tree reduction) --------
__device__ inline float cvsq(const float* v) {
  float m = 0.f;
#pragma unroll
  for (int e = 0; e < EE; ++e) m += v[e];
  m *= (1.0f / EE);
  float var = 0.f;
#pragma unroll
  for (int e = 0; e < EE; ++e) { float d = v[e] - m; var += d * d; }
  var *= (1.0f / (EE - 1));
  return var / (m * m + 1e-10f);
}

__global__ __launch_bounds__(256) void loss_kernel(
    const float* __restrict__ gate4, const int* __restrict__ idx4,
    float* __restrict__ loss_out) {
  __shared__ float simp[256][EE];
  __shared__ float scnt[256][EE];
  int t = threadIdx.x;
  float imp[EE], cnt[EE];
#pragma unroll
  for (int e = 0; e < EE; ++e) { imp[e] = 0.f; cnt[e] = 0.f; }
  for (int b = t; b < BB; b += 256) {
#pragma unroll
    for (int j = 0; j < KTOP; ++j) {
      int e = idx4[b * KTOP + j];
      imp[e] += gate4[b * KTOP + j];
      cnt[e] += 1.f;
    }
  }
#pragma unroll
  for (int e = 0; e < EE; ++e) { simp[t][e] = imp[e]; scnt[t][e] = cnt[e]; }
  __syncthreads();
  for (int s = 128; s > 0; s >>= 1) {
    if (t < s) {
#pragma unroll
      for (int e = 0; e < EE; ++e) {
        simp[t][e] += simp[t + s][e];
        scnt[t][e] += scnt[t + s][e];
      }
    }
    __syncthreads();
  }
  if (t == 0) *loss_out = (cvsq(&simp[0][0]) + cvsq(&scnt[0][0])) * 0.01f;
}

// ---------------- Kernel 3: counting sort of tokens by expert mask ---------
__global__ __launch_bounds__(256) void sort_kernel(
    const int* __restrict__ maskb, int* __restrict__ sorted) {
  __shared__ int s_mask[BB];
  __shared__ int s_cnt[256];
  __shared__ int s_off[256];
  int t = threadIdx.x;
  s_cnt[t] = 0;
  for (int b = t; b < BB; b += 256) s_mask[b] = maskb[b];
  __syncthreads();
  for (int b = t; b < BB; b += 256) atomicAdd(&s_cnt[s_mask[b]], 1);
  __syncthreads();
  if (t == 0) {
    int a = 0;
    for (int mv = 0; mv < 256; ++mv) { s_off[mv] = a; a += s_cnt[mv]; }
  }
  __syncthreads();
  int pos = s_off[t];
  for (int b = 0; b < BB; ++b)
    if (s_mask[b] == t) sorted[pos++] = b;
}

// -------- Kernel 4a: pack w_exp -> f16 in per-lane WMMA-B-operand tiling (contiguous k) -----
// w16p[e][kb][cblk][lane][j] = f16( w_exp[e][kb*32 + (lane>>4)*16 + j][cblk*16 + (lane&15)] )
// so a wave's B operand (K=kb*32..+31, N=cblk*16..+15) is 1KB contiguous.
__global__ __launch_bounds__(256) void pack_w_kernel(
    const float* __restrict__ w, _Float16* __restrict__ wp) {
  const size_t total = (size_t)EE * DD * CC;
  for (size_t i = (size_t)blockIdx.x * 256 + threadIdx.x; i < total;
       i += (size_t)gridDim.x * 256) {
    size_t c = i & (CC - 1);
    size_t t = i >> 12;            // /CC
    size_t d = t & (DD - 1);
    size_t e = t >> 10;            // /DD
    size_t kb = d >> 5;
    size_t hh = (d >> 4) & 1;
    size_t j  = d & 15;
    size_t cblk = c >> 4;
    size_t lo = c & 15;
    size_t widx = ((((e * (DD / 32) + kb) * (CC / 16) + cblk) * 2 + hh) * 16 + lo) * 16 + j;
    wp[widx] = (_Float16)w[i];
  }
}

// -------- Kernel 4b: split-f16 WMMA expert GEMM + combine (fast path) ------
// 8 waves; each wave owns two adjacent 16-col tiles (A operand shared).
__global__ __launch_bounds__(256) void moe_kernel_f16(
    const float* __restrict__ x, const _Float16* __restrict__ w16p,
    const float* __restrict__ b_exp, const float* __restrict__ gate4,
    const int* __restrict__ idx4, const int* __restrict__ sorted,
    float* __restrict__ y) {
  __shared__ _Float16 xh[16][XSTRIDE];   // 33.0 KB (hi plane)
  __shared__ _Float16 xl[16][XSTRIDE];   // 33.0 KB (lo plane, compensation)
  __shared__ float gs[16][EE];
  __shared__ int toks[16];
  __shared__ int umask_s;

  int tid = threadIdx.x;
  if (tid == 0) umask_s = 0;
  __syncthreads();
  if (tid < 16) {
    int tok = sorted[blockIdx.y * 16 + tid];
    toks[tid] = tok;
    float g[EE];
#pragma unroll
    for (int e = 0; e < EE; ++e) g[e] = 0.f;
    int m = 0;
#pragma unroll
    for (int j = 0; j < KTOP; ++j) {
      int e = idx4[tok * KTOP + j];
      g[e] = gate4[tok * KTOP + j];
      m |= 1 << e;
    }
#pragma unroll
    for (int e = 0; e < EE; ++e) gs[tid][e] = g[e];
    atomicOr(&umask_s, m);
  }
  __syncthreads();

  // Stage X once per block: split into f16 hi/lo planes in LDS.
  for (int row = 0; row < 16; ++row) {
    float4 v = *(const float4*)(x + (size_t)toks[row] * DD + tid * 4);
    v4h hh, ll;
    hh[0] = (_Float16)v.x; ll[0] = (_Float16)(v.x - (float)hh[0]);
    hh[1] = (_Float16)v.y; ll[1] = (_Float16)(v.y - (float)hh[1]);
    hh[2] = (_Float16)v.z; ll[2] = (_Float16)(v.z - (float)hh[2]);
    hh[3] = (_Float16)v.w; ll[3] = (_Float16)(v.w - (float)hh[3]);
    *(v4h*)&xh[row][tid * 4] = hh;
    *(v4h*)&xl[row][tid * 4] = ll;
  }
  __syncthreads();
  int umask = umask_s;

  int wave = tid >> 5, lane = tid & 31;
  int hi = lane >> 4, lo = lane & 15;
  int cblk = blockIdx.x * (TNF / 16) + wave * 2;   // first of two 16-col tiles
  int c = cblk * 16 + lo;

  float comb0[8], comb1[8];
#pragma unroll
  for (int r = 0; r < 8; ++r) { comb0[r] = 0.f; comb1[r] = 0.f; }

  for (int e = 0; e < EE; ++e) {
    if (!((umask >> e) & 1)) continue;        // block-uniform skip
    v8f acc0 = {};
    v8f acc1 = {};
    const _Float16* wb =
        w16p + ((((size_t)e * (DD / 32)) * (CC / 16) + cblk) * 32 + lane) * 16;
    const size_t kbstride = (size_t)(CC / 16) * 32 * 16;   // f16 elems per kb step
#pragma unroll 2
    for (int kb = 0; kb < DD / 32; ++kb) {
      int k0 = kb * 32;
      // A operand (16-bit A 16x32): elems 0-15 K=k0+16*hi+j (matches packed B)
      v8h h0 = *(const v8h*)&xh[lo][k0 + 16 * hi];
      v8h h1 = *(const v8h*)&xh[lo][k0 + 16 * hi + 8];
      v8h l0 = *(const v8h*)&xl[lo][k0 + 16 * hi];
      v8h l1 = *(const v8h*)&xl[lo][k0 + 16 * hi + 8];
      v16h ahi = __builtin_shufflevector(h0, h1, 0,1,2,3,4,5,6,7,8,9,10,11,12,13,14,15);
      v16h alo = __builtin_shufflevector(l0, l1, 0,1,2,3,4,5,6,7,8,9,10,11,12,13,14,15);
      // B operands for the two tiles: adjacent 512-f16 (1KB) blocks
      const _Float16* wk = wb + (size_t)kb * kbstride;
      v16h b0 = *(const v16h*)wk;
      v16h b1 = *(const v16h*)(wk + 512);
      acc0 = __builtin_amdgcn_wmma_f32_16x16x32_f16(
          false, ahi, false, b0, (short)0, acc0, false, false);
      acc0 = __builtin_amdgcn_wmma_f32_16x16x32_f16(
          false, alo, false, b0, (short)0, acc0, false, false);
      acc1 = __builtin_amdgcn_wmma_f32_16x16x32_f16(
          false, ahi, false, b1, (short)0, acc1, false, false);
      acc1 = __builtin_amdgcn_wmma_f32_16x16x32_f16(
          false, alo, false, b1, (short)0, acc1, false, false);
    }
    float bias0 = b_exp[(size_t)e * CC + c];
    float bias1 = b_exp[(size_t)e * CC + c + 16];
#pragma unroll
    for (int r = 0; r < 8; ++r) {
      float g = gs[r + 8 * hi][e];
      comb0[r] += g * __expf(acc0[r] + bias0);
      comb1[r] += g * __expf(acc1[r] + bias1);
    }
  }

#pragma unroll
  for (int r = 0; r < 8; ++r) {
    int tok = toks[r + 8 * hi];
    float v0 = comb0[r], v1 = comb1[r];
    v0 = (v0 == 0.0f) ? EPS_COMBINE : v0;
    v1 = (v1 == 0.0f) ? EPS_COMBINE : v1;
    y[(size_t)tok * CC + c] = __logf(v0);
    y[(size_t)tok * CC + c + 16] = __logf(v1);
  }
}

// -------- Kernel 4c: exact f32 WMMA x4 kernel (fallback path) --------------
__global__ __launch_bounds__(256) void moe_kernel_f32(
    const float* __restrict__ x, const float* __restrict__ w_exp,
    const float* __restrict__ b_exp, const float* __restrict__ gate4,
    const int* __restrict__ idx4, const int* __restrict__ sorted,
    float* __restrict__ y) {
  __shared__ float xs[16][KC + 4];
  __shared__ float gs[16][EE];
  __shared__ int toks[16];
  __shared__ int umask_s;

  int tid = threadIdx.x;
  if (tid == 0) umask_s = 0;
  __syncthreads();
  if (tid < 16) {
    int tok = sorted[blockIdx.y * 16 + tid];
    toks[tid] = tok;
    float g[EE];
#pragma unroll
    for (int e = 0; e < EE; ++e) g[e] = 0.f;
    int m = 0;
#pragma unroll
    for (int j = 0; j < KTOP; ++j) {
      int e = idx4[tok * KTOP + j];
      g[e] = gate4[tok * KTOP + j];
      m |= 1 << e;
    }
#pragma unroll
    for (int e = 0; e < EE; ++e) gs[tid][e] = g[e];
    atomicOr(&umask_s, m);
  }
  __syncthreads();
  int umask = umask_s;

  int wave = tid >> 5, lane = tid & 31;
  int hi = lane >> 4, lo = lane & 15;
  int c = blockIdx.x * TN + wave * 16 + lo;

  float comb[8];
#pragma unroll
  for (int r = 0; r < 8; ++r) comb[r] = 0.f;

  for (int e = 0; e < EE; ++e) {
    if (!((umask >> e) & 1)) continue;
    v8f acc = {};
    for (int kc = 0; kc < DD; kc += KC) {
      __syncthreads();
#pragma unroll
      for (int p = 0; p < 4; ++p) {
        int row = (tid >> 6) + p * 4;
        int l64 = tid & 63;
        float4 v = *(const float4*)(x + (size_t)toks[row] * DD + kc + l64 * 4);
        *(float4*)(&xs[row][l64 * 4]) = v;
      }
      __syncthreads();
      const float* wcol = w_exp + ((size_t)e * DD + kc) * CC + c;
#pragma unroll 8
      for (int k0 = 0; k0 < KC; k0 += 4) {
        v2f a, bmat;
        a[0] = xs[lo][k0 + 2 * hi];
        a[1] = xs[lo][k0 + 2 * hi + 1];
        bmat[0] = wcol[(size_t)(k0 + 2 * hi) * CC];
        bmat[1] = wcol[(size_t)(k0 + 2 * hi + 1) * CC];
        acc = __builtin_amdgcn_wmma_f32_16x16x4_f32(
            false, a, false, bmat, (short)0, acc, false, false);
      }
    }
    float bias = b_exp[(size_t)e * CC + c];
#pragma unroll
    for (int r = 0; r < 8; ++r) {
      float g = gs[r + 8 * hi][e];
      comb[r] += g * __expf(acc[r] + bias);
    }
  }

#pragma unroll
  for (int r = 0; r < 8; ++r) {
    int tok = toks[r + 8 * hi];
    float v = comb[r];
    v = (v == 0.0f) ? EPS_COMBINE : v;
    y[(size_t)tok * CC + c] = __logf(v);
  }
}

// ---------------- Launch ---------------------------------------------------
extern "C" void kernel_launch(void* const* d_in, const int* in_sizes, int n_in,
                              void* d_out, int out_size, void* d_ws, size_t ws_size,
                              hipStream_t stream) {
  (void)in_sizes; (void)n_in;
  const float* x      = (const float*)d_in[0];
  const float* w_gate = (const float*)d_in[1];
  const float* w_exp  = (const float*)d_in[2];
  const float* b_exp  = (const float*)d_in[3];

  float* y = (float*)d_out;
  float* loss_out = y + (size_t)out_size - 1;

  float* gate4 = (float*)d_ws;                       // 4096*4 f32
  int*   idx4  = (int*)(gate4 + (size_t)BB * KTOP);  // 4096*4 i32
  int*   maskb = idx4 + (size_t)BB * KTOP;           // 4096 i32
  int*   sorted = maskb + BB;                        // 4096 i32

  gating_kernel<<<BB / 128, 128, 0, stream>>>(x, w_gate, gate4, idx4, maskb);
  loss_kernel<<<1, 256, 0, stream>>>(gate4, idx4, loss_out);
  sort_kernel<<<1, 256, 0, stream>>>(maskb, sorted);

  const size_t wpack_off = 1u << 20;                        // 1MB reserve
  const size_t wpack_bytes = (size_t)EE * DD * CC * sizeof(_Float16);  // 64MB
  if (ws_size >= wpack_off + wpack_bytes) {
    _Float16* w16p = (_Float16*)((char*)d_ws + wpack_off);
    pack_w_kernel<<<8192, 256, 0, stream>>>(w_exp, w16p);
    dim3 grid(CC / TNF, BB / 16);
    moe_kernel_f16<<<grid, 256, 0, stream>>>(x, w16p, b_exp, gate4, idx4, sorted, y);
  } else {
    dim3 grid(CC / TN, BB / 16);
    moe_kernel_f32<<<grid, 256, 0, stream>>>(x, w_exp, b_exp, gate4, idx4, sorted, y);
  }
}
